// GatedCrossAttention_77429670412499
// MI455X (gfx1250) — matrix operand order repaired and checked by reading, not compiled, not run on hardware
//
#include <hip/hip_runtime.h>
#include <hip/hip_bf16.h>
#include <math.h>

typedef _Float16 half_t;
typedef __attribute__((ext_vector_type(16))) _Float16 v16h;
typedef __attribute__((ext_vector_type(8)))  _Float16 v8h;
typedef __attribute__((ext_vector_type(8)))  float    v8f;

#define B_   16
#define LQ_  512
#define LK_  1024
#define DM_  512
#define H_   8
#define DH_  64

union V16U { v16h v; v8h h[2]; };

__device__ __forceinline__ v8f wmma_f16(v16h a, v16h b, v8f c) {
  // D = A(16x32 f16) * B(32x16 f16) + C(16x16 f32)
  return __builtin_amdgcn_wmma_f32_16x16x32_f16(false, a, false, b, (short)0, c,
                                                false, false);
}

__device__ __forceinline__ float rmax16(float x) {
  #pragma unroll
  for (int m = 1; m < 16; m <<= 1) x = fmaxf(x, __shfl_xor(x, m, 32));
  return x;
}
__device__ __forceinline__ float rsum16(float x) {
  #pragma unroll
  for (int m = 1; m < 16; m <<= 1) x += __shfl_xor(x, m, 32);
  return x;
}

// Raw hardware transcendentals (v_exp_f32 / v_log_f32 are base-2).
// Inputs here are well-conditioned: gate in (1e-3,1), exp args <= 0.
__device__ __forceinline__ float fast_exp(float x) {
  return __builtin_amdgcn_exp2f(x * 1.44269504088896f);
}
__device__ __forceinline__ float fast_ln(float x) {
  return __builtin_amdgcn_logf(x) * 0.693147180559945f;
}

// Async global -> LDS copy, 16 bytes per lane, tracked on ASYNCcnt.
__device__ __forceinline__ void async_g2l_b128(unsigned lds_byte_off,
                                               const half_t* gaddr) {
  asm volatile("global_load_async_to_lds_b128 %0, %1, off"
               :: "v"(lds_byte_off), "v"(gaddr) : "memory");
}
__device__ __forceinline__ void wait_async0() {
  asm volatile("s_wait_asynccnt 0x0" ::: "memory");
}
__device__ __forceinline__ unsigned lds_off(const void* p) {
  return (unsigned)(unsigned long long)(__attribute__((address_space(3))) const void*)p;
}

// ---------------------------------------------------------------- converts
__global__ void cvt_f32_to_f16(const float* __restrict__ in,
                               half_t* __restrict__ out, int n) {
  int i = blockIdx.x * blockDim.x + threadIdx.x;
  if (i < n) out[i] = (half_t)in[i];
}

// W[k][n] (512x512 f32) -> Wt[n][k] f16 (so WMMA B-fragments are contiguous)
__global__ void cvt_transpose512(const float* __restrict__ in,
                                 half_t* __restrict__ out) {
  int i = blockIdx.x * blockDim.x + threadIdx.x;   // 0 .. 512*512-1
  int k = i >> 9;
  int n = i & 511;
  out[n * 512 + k] = (half_t)in[i];
}

// ---------------------------------------------------------------- GEMM
// X[M x 512] (f16 row-major) @ Wt (stored [n][k] f16) + bias -> out [M x 512]
// block = 8 waves; each wave computes a 32x32 tile (2x2 WMMA accum tiles).
template <bool OUT_F32>
__global__ __launch_bounds__(256) void gemm_wmma(
    const half_t* __restrict__ X, const half_t* __restrict__ Wt,
    const float* __restrict__ bias, half_t* __restrict__ Oh,
    float* __restrict__ Of) {
  const int lane = threadIdx.x;
  const int wave = threadIdx.y;
  const int l15  = lane & 15;
  const int hi   = lane >> 4;
  const int mbase = blockIdx.x * 64  + (wave >> 2) * 32;
  const int nbase = blockIdx.y * 128 + (wave & 3)  * 32;

  const half_t* a0 = X  + (size_t)(mbase + l15) * DM_;
  const half_t* a1 = a0 + (size_t)16 * DM_;
  const half_t* b0 = Wt + (size_t)(nbase + l15) * DM_;
  const half_t* b1 = b0 + (size_t)16 * DM_;

  v8f acc00 = {}, acc01 = {}, acc10 = {}, acc11 = {};

  #pragma unroll 4
  for (int kk = 0; kk < DM_; kk += 32) {
    // pull next K-chunk's lines toward the WGP while WMMAs run
    if (kk + 128 < DM_) {
      __builtin_prefetch(a0 + kk + 128, 0, 0);
      __builtin_prefetch(a1 + kk + 128, 0, 0);
      __builtin_prefetch(b0 + kk + 128, 0, 0);
      __builtin_prefetch(b1 + kk + 128, 0, 0);
    }
    V16U af0, af1;
    // A k order used here (same as B): lane holds K = e + 16*hi  (+kk)
    af0.h[0] = *(const v8h*)(a0 + kk +     16 * hi);
    af0.h[1] = *(const v8h*)(a0 + kk + 8 + 16 * hi);
    af1.h[0] = *(const v8h*)(a1 + kk +     16 * hi);
    af1.h[1] = *(const v8h*)(a1 + kk + 8 + 16 * hi);
    // B k order, same as A: lane holds K = e + 16*hi  (contiguous in Wt[n][k])
    v16h bf0 = *(const v16h*)(b0 + kk + 16 * hi);
    v16h bf1 = *(const v16h*)(b1 + kk + 16 * hi);
    acc00 = wmma_f16(af0.v, bf0, acc00);
    acc01 = wmma_f16(af0.v, bf1, acc01);
    acc10 = wmma_f16(af1.v, bf0, acc10);
    acc11 = wmma_f16(af1.v, bf1, acc11);
  }

  // C/D layout: vgpr r -> row r + 8*hi ; col = l15
  #pragma unroll
  for (int r = 0; r < 8; ++r) {
    const int m0 = mbase + r + 8 * hi;
    const int m1 = m0 + 16;
    const int c0 = nbase + l15;
    const int c1 = c0 + 16;
    const float v00 = acc00[r] + bias[c0];
    const float v01 = acc01[r] + bias[c1];
    const float v10 = acc10[r] + bias[c0];
    const float v11 = acc11[r] + bias[c1];
    if (OUT_F32) {
      Of[(size_t)m0 * DM_ + c0] = v00;
      Of[(size_t)m0 * DM_ + c1] = v01;
      Of[(size_t)m1 * DM_ + c0] = v10;
      Of[(size_t)m1 * DM_ + c1] = v11;
    } else {
      Oh[(size_t)m0 * DM_ + c0] = (half_t)v00;
      Oh[(size_t)m0 * DM_ + c1] = (half_t)v01;
      Oh[(size_t)m1 * DM_ + c0] = (half_t)v10;
      Oh[(size_t)m1 * DM_ + c1] = (half_t)v11;
    }
  }
}

// ---------------------------------------------------------------- attention
// grid = (B*H, LQ/128), block = (32,8). Each wave: 16 queries, flash-style
// online softmax over LK in 32-key chunks. K tile staged with async-to-LDS.
__global__ __launch_bounds__(256) void attn_wmma(
    const half_t* __restrict__ Qp, const half_t* __restrict__ Kp,
    const half_t* __restrict__ Vp, const float* __restrict__ gate,
    const int* __restrict__ mask, half_t* __restrict__ ctx) {
  const int bh = blockIdx.x;
  const int b  = bh >> 3;
  const int h  = bh & 7;
  const int wave = threadIdx.y;
  const int lane = threadIdx.x;
  const int l15  = lane & 15;
  const int hi   = lane >> 4;
  const int qbase = blockIdx.y * 128 + wave * 16;
  const int tid   = wave * 32 + lane;

  __shared__ half_t Ktile[32 * 64];      // [key][d]
  __shared__ half_t Vtile[64 * 32];      // [d][key]  (transposed)
  __shared__ half_t Ptile[8][16 * 32];   // per-wave [q][key]

  const float SC   = 0.15625f;           // (1/sqrt(64)) / 0.8
  const float invT = 1.25f;              // 1/0.8
  const float NEGT = -1.25e9f;           // -1e9 / 0.8

  const float* gate_b = gate + (size_t)b * LK_;
  const int*   mask_b = mask + (size_t)b * LK_;

  // Load the wave's Q block (16 x 64) as two A-fragments, kept in registers.
  const half_t* qrow = Qp + (size_t)(b * LQ_ + qbase + l15) * DM_ + h * DH_;
  V16U qa[2];
  #pragma unroll
  for (int c = 0; c < 2; ++c) {
    qa[c].h[0] = *(const v8h*)(qrow + 32 * c +     16 * hi);
    qa[c].h[1] = *(const v8h*)(qrow + 32 * c + 8 + 16 * hi);
  }

  v8f o0 = {}, o1 = {}, o2 = {}, o3 = {};
  float m_r[8], l_r[8];
  #pragma unroll
  for (int r = 0; r < 8; ++r) { m_r[r] = -1e30f; l_r[r] = 0.0f; }

  half_t* pw = Ptile[wave];

  for (int kb = 0; kb < LK_; kb += 32) {
    __syncthreads();   // previous chunk's K/V consumers done
    {
      // Stage 32 keys x 64 dims. K: async global->LDS (no VGPR round trip).
      // V: VGPR round trip with on-the-fly transpose to [d][key].
      const int key   = tid >> 3;
      const int dbase = (tid & 7) * 8;
      const size_t row = (size_t)(b * LK_ + kb + key) * DM_ + h * DH_ + dbase;
      async_g2l_b128(lds_off(Ktile + key * 64 + dbase), Kp + row);
      v8h vv = *(const v8h*)(Vp + row);
      #pragma unroll
      for (int t = 0; t < 8; ++t) Vtile[(dbase + t) * 32 + key] = vv[t];
      if (kb + 32 < LK_) {
        __builtin_prefetch(Kp + row + (size_t)32 * DM_, 0, 0);
        __builtin_prefetch(Vp + row + (size_t)32 * DM_, 0, 0);
      }
      wait_async0();
    }
    __syncthreads();

    // S = Q @ K^T : two 16x16 key tiles, contraction Dh=64 in two steps.
    v8f s0 = {}, s1 = {};
    #pragma unroll
    for (int c = 0; c < 2; ++c) {
      v16h bk0 = *(const v16h*)(Ktile + l15        * 64 + 32 * c + 16 * hi);
      v16h bk1 = *(const v16h*)(Ktile + (l15 + 16) * 64 + 32 * c + 16 * hi);
      s0 = wmma_f16(qa[c].v, bk0, s0);
      s1 = wmma_f16(qa[c].v, bk1, s1);
    }

    // Gate / mask biases, per-column (one key per lane per tile).
    const int key0 = kb + l15;
    const int key1 = key0 + 16;
    const float g0 = fast_ln(gate_b[key0]) * invT;
    const float g1 = fast_ln(gate_b[key1]) * invT;
    const bool  ok0 = mask_b[key0] != 0;
    const bool  ok1 = mask_b[key1] != 0;

    // Online softmax, rows r + 8*hi, reductions across 16-lane halves.
    #pragma unroll
    for (int r = 0; r < 8; ++r) {
      float z0 = ok0 ? s0[r] * SC + g0 : NEGT;
      float z1 = ok1 ? s1[r] * SC + g1 : NEGT;
      float mx = rmax16(fmaxf(z0, z1));
      float mnew = fmaxf(m_r[r], mx);
      float corr = fast_exp(m_r[r] - mnew);
      float p0 = fast_exp(z0 - mnew);
      float p1 = fast_exp(z1 - mnew);
      l_r[r] = l_r[r] * corr + rsum16(p0 + p1);
      m_r[r] = mnew;
      o0[r] *= corr; o1[r] *= corr; o2[r] *= corr; o3[r] *= corr;
      const int row = r + 8 * hi;
      pw[row * 32 + l15]      = (half_t)p0;
      pw[row * 32 + 16 + l15] = (half_t)p1;
    }

    // Wave-local LDS RAW: ds ops are in-order per wave; stop compiler reorder.
    __asm__ volatile("" ::: "memory");

    // P (16x32) as A-fragment, V^T rows as B-fragments, O += P @ V.
    V16U pa;
    pa.h[0] = *(const v8h*)(pw + l15 * 32 +     16 * hi);
    pa.h[1] = *(const v8h*)(pw + l15 * 32 + 8 + 16 * hi);
    v16h vb0 = *(const v16h*)(Vtile + (l15)      * 32 + 16 * hi);
    v16h vb1 = *(const v16h*)(Vtile + (16 + l15) * 32 + 16 * hi);
    v16h vb2 = *(const v16h*)(Vtile + (32 + l15) * 32 + 16 * hi);
    v16h vb3 = *(const v16h*)(Vtile + (48 + l15) * 32 + 16 * hi);
    o0 = wmma_f16(pa.v, vb0, o0);
    o1 = wmma_f16(pa.v, vb1, o1);
    o2 = wmma_f16(pa.v, vb2, o2);
    o3 = wmma_f16(pa.v, vb3, o3);
  }

  // Normalize and store ctx (f16, layout [B, LQ, H*Dh]).
  #pragma unroll
  for (int r = 0; r < 8; ++r) {
    const float inv = 1.0f / l_r[r];
    const int q = qbase + r + 8 * hi;
    const size_t base = (size_t)(b * LQ_ + q) * DM_ + h * DH_;
    ctx[base + l15]      = (half_t)(o0[r] * inv);
    ctx[base + 16 + l15] = (half_t)(o1[r] * inv);
    ctx[base + 32 + l15] = (half_t)(o2[r] * inv);
    ctx[base + 48 + l15] = (half_t)(o3[r] * inv);
  }
}

// ---------------------------------------------------------------- launcher
extern "C" void kernel_launch(void* const* d_in, const int* in_sizes, int n_in,
                              void* d_out, int out_size, void* d_ws,
                              size_t ws_size, hipStream_t stream) {
  const float* Q_src = (const float*)d_in[0];
  const float* KV_src = (const float*)d_in[1];
  const float* key_gate = (const float*)d_in[2];
  const int*   key_mask = (const int*)d_in[3];
  const float* Wq = (const float*)d_in[4];
  const float* bq = (const float*)d_in[5];
  const float* Wk = (const float*)d_in[6];
  const float* bk = (const float*)d_in[7];
  const float* Wv = (const float*)d_in[8];
  const float* bv = (const float*)d_in[9];
  const float* Wo = (const float*)d_in[10];
  const float* bo = (const float*)d_in[11];
  float* out = (float*)d_out;

  const size_t NQ = (size_t)B_ * LQ_ * DM_;   // 4,194,304
  const size_t NK = (size_t)B_ * LK_ * DM_;   // 8,388,608
  const size_t NW = (size_t)DM_ * DM_;        //   262,144

  half_t* ws = (half_t*)d_ws;
  half_t* Xq  = ws;          // f16 Q_src
  half_t* Xkv = Xq  + NQ;    // f16 KV_src
  half_t* Wqt = Xkv + NK;    // transposed f16 weights
  half_t* Wkt = Wqt + NW;
  half_t* Wvt = Wkt + NW;
  half_t* Wot = Wvt + NW;
  half_t* Qp  = Wot + NW;    // projected q
  half_t* Kp  = Qp  + NQ;    // projected k
  half_t* Vp  = Kp  + NK;    // projected v
  half_t* Ctx = Vp  + NK;    // attention context

  // 1) convert activations to f16
  cvt_f32_to_f16<<<(int)(NQ / 256), 256, 0, stream>>>(Q_src, Xq, (int)NQ);
  cvt_f32_to_f16<<<(int)(NK / 256), 256, 0, stream>>>(KV_src, Xkv, (int)NK);

  // 2) convert + transpose weights to f16
  cvt_transpose512<<<(int)(NW / 256), 256, 0, stream>>>(Wq, Wqt);
  cvt_transpose512<<<(int)(NW / 256), 256, 0, stream>>>(Wk, Wkt);
  cvt_transpose512<<<(int)(NW / 256), 256, 0, stream>>>(Wv, Wvt);
  cvt_transpose512<<<(int)(NW / 256), 256, 0, stream>>>(Wo, Wot);

  dim3 blk(32, 8);
  // 3) Q/K/V projections (WMMA GEMMs, f16 out)
  gemm_wmma<false><<<dim3(B_ * LQ_ / 64, DM_ / 128), blk, 0, stream>>>(
      Xq, Wqt, bq, Qp, nullptr);
  gemm_wmma<false><<<dim3(B_ * LK_ / 64, DM_ / 128), blk, 0, stream>>>(
      Xkv, Wkt, bk, Kp, nullptr);
  gemm_wmma<false><<<dim3(B_ * LK_ / 64, DM_ / 128), blk, 0, stream>>>(
      Xkv, Wvt, bv, Vp, nullptr);

  // 4) fused gated-masked flash attention
  attn_wmma<<<dim3(B_ * H_, LQ_ / 128), blk, 0, stream>>>(
      Qp, Kp, Vp, key_gate, key_mask, Ctx);

  // 5) output projection (f32 out + bias)
  gemm_wmma<true><<<dim3(B_ * LQ_ / 64, DM_ / 128), blk, 0, stream>>>(
      Ctx, Wot, bo, nullptr, out);
}
